// FlashDeepseekLayer_89773406421359
// MI455X (gfx1250) — matrix operand order repaired and checked by reading, not compiled, not run on hardware
//
#include <hip/hip_runtime.h>
#include <hip/hip_bf16.h>

#define T_TOK 2048
#define H_DIM 2048
#define N_EXP 16
#define K_TOP 4
#define I_DIM 1408
#define IS_DIM 2816
#define CAP_N 1024

typedef __attribute__((ext_vector_type(16))) __bf16 bf16x16;
typedef __attribute__((ext_vector_type(8)))  __bf16 bf16x8;
typedef __attribute__((ext_vector_type(8)))  float  f32x8;

__device__ __forceinline__ bf16x16 join16(bf16x8 lo, bf16x8 hi) {
  return __builtin_shufflevector(lo, hi, 0, 1, 2, 3, 4, 5, 6, 7,
                                 8, 9, 10, 11, 12, 13, 14, 15);
}

__device__ __forceinline__ unsigned lds_addr32(const void* p) {
  // Flat->LDS mapping uses addr[31:0] (workgroup-relative); truncate.
  return (unsigned)(unsigned long long)p;
}

// ---------------------------------------------------------------------------
// Gate: logits = x @ gate_w^T -> softmax -> top-4 (stable) -> normalized w
// ---------------------------------------------------------------------------
__global__ __launch_bounds__(128) void gate_topk_kernel(
    const float* __restrict__ X, const float* __restrict__ GW,
    int* __restrict__ tki, float* __restrict__ tkw) {
  __shared__ float red[128 * N_EXP];
  const int t = blockIdx.x, tid = threadIdx.x;
  float acc[N_EXP];
#pragma unroll
  for (int e = 0; e < N_EXP; ++e) acc[e] = 0.f;
  for (int h = tid; h < H_DIM; h += 128) {
    float xv = X[(long)t * H_DIM + h];
#pragma unroll
    for (int e = 0; e < N_EXP; ++e) acc[e] += xv * GW[(long)e * H_DIM + h];
  }
#pragma unroll
  for (int e = 0; e < N_EXP; ++e) red[tid * N_EXP + e] = acc[e];
  __syncthreads();
  for (int s = 64; s > 0; s >>= 1) {
    if (tid < s) {
#pragma unroll
      for (int e = 0; e < N_EXP; ++e)
        red[tid * N_EXP + e] += red[(tid + s) * N_EXP + e];
    }
    __syncthreads();
  }
  if (tid == 0) {
    float sc[N_EXP], mx = -1e30f;
#pragma unroll
    for (int e = 0; e < N_EXP; ++e) { sc[e] = red[e]; mx = fmaxf(mx, sc[e]); }
    float sum = 0.f;
#pragma unroll
    for (int e = 0; e < N_EXP; ++e) { sc[e] = __expf(sc[e] - mx); sum += sc[e]; }
    float inv = 1.f / sum;
#pragma unroll
    for (int e = 0; e < N_EXP; ++e) sc[e] *= inv;
    int ti[K_TOP]; float tw[K_TOP]; float wsum = 0.f;
    bool used[N_EXP];
#pragma unroll
    for (int e = 0; e < N_EXP; ++e) used[e] = false;
#pragma unroll
    for (int k = 0; k < K_TOP; ++k) {
      int bi = 0; float bv = -1.f;
      for (int e = 0; e < N_EXP; ++e)
        if (!used[e] && sc[e] > bv) { bv = sc[e]; bi = e; }  // strict > == first index
      used[bi] = true; ti[k] = bi; tw[k] = bv; wsum += bv;
    }
    float nrm = 1.f / (wsum + 1e-20f);
#pragma unroll
    for (int k = 0; k < K_TOP; ++k) {
      tki[t * K_TOP + k] = ti[k];
      tkw[t * K_TOP + k] = tw[k] * nrm;
    }
  }
}

// ---------------------------------------------------------------------------
// Dispatch: ballot-prefix rank per expert == reference stable sort + capacity
// ---------------------------------------------------------------------------
__global__ __launch_bounds__(32) void dispatch_kernel(
    const int* __restrict__ tki, const float* __restrict__ tkw,
    int* __restrict__ buf, float* __restrict__ wbuf) {
  const int e = blockIdx.x, lane = threadIdx.x;
  int rank = 0;
  for (int base = 0; base < T_TOK * K_TOP; base += 32) {
    int s = base + lane;
    bool m = (tki[s] == e);
    unsigned mask = __builtin_amdgcn_ballot_w32(m);
    int r = rank + __builtin_popcount(mask & ((1u << lane) - 1u));
    if (m && r < CAP_N) {
      buf[e * CAP_N + r] = s >> 2;
      wbuf[e * CAP_N + r] = tkw[s];
    }
    rank += __builtin_popcount(mask);
  }
  int rmin = rank < CAP_N ? rank : CAP_N;
  for (int r = rmin + lane; r < CAP_N; r += 32) {
    buf[e * CAP_N + r] = T_TOK;   // sentinel -> zero row, dropped on scatter
    wbuf[e * CAP_N + r] = 0.f;
  }
}

// ---------------------------------------------------------------------------
// GEMM1 (SwiGLU): Mid[e] = silu(Xg @ Wg[e]) * (Xg @ Wu[e]), bf16 out.
// Block 128(M) x 64(N), K step 32 over H. Reg-pipelined global loads.
// ---------------------------------------------------------------------------
__global__ __launch_bounds__(256) void gemm1_swiglu_kernel(
    const float* __restrict__ X, const int* __restrict__ buf,
    const float* __restrict__ Wg, const float* __restrict__ Wu,
    __bf16* __restrict__ Mid, int Ndim, int Mrows) {
  __shared__ alignas(16) __bf16 As[128 * 40];
  __shared__ alignas(16) __bf16 Bgs[64 * 40];
  __shared__ alignas(16) __bf16 Bus[64 * 40];
  __shared__ int rowTok[128];

  const int tid = threadIdx.x;
  const int n0 = blockIdx.x * 64;
  const int m0 = blockIdx.y * 128;
  const int e = blockIdx.z;
  const float* Wgb = Wg + (long)e * H_DIM * Ndim;
  const float* Wub = Wu + (long)e * H_DIM * Ndim;

  if (tid < 128) rowTok[tid] = buf ? buf[e * CAP_N + m0 + tid] : (m0 + tid);
  __syncthreads();

  const int lane = tid & 31, wid = tid >> 5;
  const int half = lane >> 4, l15 = lane & 15;
  const int wm = (wid & 3) * 32, wn = (wid >> 2) * 32;

  const f32x8 zacc = {0.f, 0.f, 0.f, 0.f, 0.f, 0.f, 0.f, 0.f};
  f32x8 accg[2][2], accu[2][2];
#pragma unroll
  for (int a = 0; a < 2; ++a)
#pragma unroll
    for (int b = 0; b < 2; ++b) { accg[a][b] = zacc; accu[a][b] = zacc; }

  float4 ra[4], rbg[2], rbu[2];

  auto loadTile = [&](int k0) {
#pragma unroll
    for (int i = 0; i < 4; ++i) {
      int c = tid + i * 256;
      int row = c >> 3, q = c & 7;
      int tok = rowTok[row];
      float4 v = {0.f, 0.f, 0.f, 0.f};
      if (tok < T_TOK) v = *(const float4*)(X + (long)tok * H_DIM + k0 + q * 4);
      ra[i] = v;
    }
#pragma unroll
    for (int i = 0; i < 2; ++i) {
      int c = tid + i * 256;
      int row = c >> 4, q = c & 15;
      rbg[i] = *(const float4*)(Wgb + (long)(k0 + row) * Ndim + n0 + q * 4);
      rbu[i] = *(const float4*)(Wub + (long)(k0 + row) * Ndim + n0 + q * 4);
    }
  };
  auto storeTile = [&]() {
#pragma unroll
    for (int i = 0; i < 4; ++i) {
      int c = tid + i * 256;
      int row = c >> 3, q = c & 7;
      __bf16* dst = &As[row * 40 + q * 4];
      dst[0] = (__bf16)ra[i].x; dst[1] = (__bf16)ra[i].y;
      dst[2] = (__bf16)ra[i].z; dst[3] = (__bf16)ra[i].w;
    }
#pragma unroll
    for (int i = 0; i < 2; ++i) {
      int c = tid + i * 256;
      int row = c >> 4, q = c & 15;
      Bgs[(q * 4 + 0) * 40 + row] = (__bf16)rbg[i].x;
      Bgs[(q * 4 + 1) * 40 + row] = (__bf16)rbg[i].y;
      Bgs[(q * 4 + 2) * 40 + row] = (__bf16)rbg[i].z;
      Bgs[(q * 4 + 3) * 40 + row] = (__bf16)rbg[i].w;
      Bus[(q * 4 + 0) * 40 + row] = (__bf16)rbu[i].x;
      Bus[(q * 4 + 1) * 40 + row] = (__bf16)rbu[i].y;
      Bus[(q * 4 + 2) * 40 + row] = (__bf16)rbu[i].z;
      Bus[(q * 4 + 3) * 40 + row] = (__bf16)rbu[i].w;
    }
  };

  loadTile(0);
  for (int k0 = 0; k0 < H_DIM; k0 += 32) {
    storeTile();
    if (k0 + 32 < H_DIM) loadTile(k0 + 32);  // overlap HBM with WMMA
    __syncthreads();

    bf16x16 af[2];
#pragma unroll
    for (int tm = 0; tm < 2; ++tm) {
      const __bf16* p = &As[(wm + tm * 16 + l15) * 40 + half * 16];
      af[tm] = join16(*(const bf16x8*)p, *(const bf16x8*)(p + 8));
    }
#pragma unroll
    for (int tn = 0; tn < 2; ++tn) {
      const __bf16* pg = &Bgs[(wn + tn * 16 + l15) * 40 + half * 16];
      bf16x16 bg = join16(*(const bf16x8*)pg, *(const bf16x8*)(pg + 8));
      const __bf16* pu = &Bus[(wn + tn * 16 + l15) * 40 + half * 16];
      bf16x16 bu = join16(*(const bf16x8*)pu, *(const bf16x8*)(pu + 8));
#pragma unroll
      for (int tm = 0; tm < 2; ++tm) {
        accg[tm][tn] = __builtin_amdgcn_wmma_f32_16x16x32_bf16(
            false, af[tm], false, bg, (short)0, accg[tm][tn], false, false);
        accu[tm][tn] = __builtin_amdgcn_wmma_f32_16x16x32_bf16(
            false, af[tm], false, bu, (short)0, accu[tm][tn], false, false);
      }
    }
    __syncthreads();
  }

  const long midBase = (long)e * Mrows * Ndim;
#pragma unroll
  for (int tm = 0; tm < 2; ++tm)
#pragma unroll
    for (int tn = 0; tn < 2; ++tn)
#pragma unroll
      for (int r = 0; r < 8; ++r) {
        int lrow = wm + tm * 16 + half * 8 + r;
        int col = n0 + wn + tn * 16 + l15;
        float g = accg[tm][tn][r];
        float u = accu[tm][tn][r];
        float s = g / (1.f + __expf(-g));
        Mid[midBase + (long)(m0 + lrow) * Ndim + col] = (__bf16)(s * u);
      }
}

// ---------------------------------------------------------------------------
// GEMM2 (down-proj): out = (Mid @ Wd[e]) * w. A-tile is a straight bf16 copy:
// use async global->LDS (ASYNCcnt) with a double-buffered LDS tile; B-tile
// (fp32->bf16 transpose) is reg-pipelined.
// ---------------------------------------------------------------------------
__global__ __launch_bounds__(256) void gemm2_down_kernel(
    const __bf16* __restrict__ Mid, const float* __restrict__ Wd,
    const int* __restrict__ buf, const float* __restrict__ wbuf,
    float* __restrict__ Y, int Kdim, int Mrows) {
  __shared__ alignas(16) __bf16 As[2 * 128 * 40];  // double buffer
  __shared__ alignas(16) __bf16 Bs[128 * 40];
  __shared__ int rowTok[128];
  __shared__ float rowW[128];

  const int tid = threadIdx.x;
  const int n0 = blockIdx.x * 128;
  const int m0 = blockIdx.y * 128;
  const int e = blockIdx.z;
  const __bf16* MidB = Mid + (long)e * Mrows * Kdim;
  const float* Wdb = Wd + (long)e * Kdim * H_DIM;

  if (tid < 128) {
    rowTok[tid] = buf ? buf[e * CAP_N + m0 + tid] : (m0 + tid);
    rowW[tid] = buf ? wbuf[e * CAP_N + m0 + tid] : 1.0f;
  }
  __syncthreads();

  const int lane = tid & 31, wid = tid >> 5;
  const int half = lane >> 4, l15 = lane & 15;
  const int wm = (wid & 3) * 32, wn = (wid >> 2) * 64;

  const f32x8 zacc = {0.f, 0.f, 0.f, 0.f, 0.f, 0.f, 0.f, 0.f};
  f32x8 acc[2][4];
#pragma unroll
  for (int a = 0; a < 2; ++a)
#pragma unroll
    for (int b = 0; b < 4; ++b) acc[a][b] = zacc;

  // Async copy of one 128x32 bf16 A tile into LDS buffer bufIdx.
  auto asyncA = [&](int k0, int bufIdx) {
#pragma unroll
    for (int i = 0; i < 2; ++i) {
      int c = tid + i * 256;
      int row = c >> 2, q = c & 3;
      const __bf16* g = MidB + (long)(m0 + row) * Kdim + k0 + q * 8;
      unsigned l = lds_addr32(&As[bufIdx * (128 * 40) + row * 40 + q * 8]);
      asm volatile("global_load_async_to_lds_b128 %0, %1, off"
                   :: "v"(l), "v"((unsigned long long)g)
                   : "memory");
    }
  };

  float4 rb[4];
  auto loadB = [&](int k0) {
#pragma unroll
    for (int i = 0; i < 4; ++i) {
      int c = tid + i * 256;
      int row = c >> 5, q = c & 31;
      rb[i] = *(const float4*)(Wdb + (long)(k0 + row) * H_DIM + n0 + q * 4);
    }
  };
  auto storeB = [&]() {
#pragma unroll
    for (int i = 0; i < 4; ++i) {
      int c = tid + i * 256;
      int row = c >> 5, q = c & 31;
      Bs[(q * 4 + 0) * 40 + row] = (__bf16)rb[i].x;
      Bs[(q * 4 + 1) * 40 + row] = (__bf16)rb[i].y;
      Bs[(q * 4 + 2) * 40 + row] = (__bf16)rb[i].z;
      Bs[(q * 4 + 3) * 40 + row] = (__bf16)rb[i].w;
    }
  };

  const int nIter = Kdim / 32;
  asyncA(0, 0);
  loadB(0);
  for (int kb = 0; kb < nIter; ++kb) {
    const int cur = kb & 1;
    storeB();
    if (kb + 1 < nIter) {
      asyncA((kb + 1) * 32, cur ^ 1);  // fill other buffer while computing
      loadB((kb + 1) * 32);
      // 2 newly-issued async ops may remain in flight; batch kb must be done.
      asm volatile("s_wait_asynccnt 0x2" ::: "memory");
    } else {
      asm volatile("s_wait_asynccnt 0x0" ::: "memory");
    }
    __syncthreads();

    const __bf16* Ab = &As[cur * (128 * 40)];
    bf16x16 af[2];
#pragma unroll
    for (int tm = 0; tm < 2; ++tm) {
      const __bf16* p = &Ab[(wm + tm * 16 + l15) * 40 + half * 16];
      af[tm] = join16(*(const bf16x8*)p, *(const bf16x8*)(p + 8));
    }
#pragma unroll
    for (int tn = 0; tn < 4; ++tn) {
      const __bf16* pb = &Bs[(wn + tn * 16 + l15) * 40 + half * 16];
      bf16x16 bf = join16(*(const bf16x8*)pb, *(const bf16x8*)(pb + 8));
#pragma unroll
      for (int tm = 0; tm < 2; ++tm)
        acc[tm][tn] = __builtin_amdgcn_wmma_f32_16x16x32_bf16(
            false, af[tm], false, bf, (short)0, acc[tm][tn], false, false);
    }
    __syncthreads();
  }

#pragma unroll
  for (int tm = 0; tm < 2; ++tm)
#pragma unroll
    for (int tn = 0; tn < 4; ++tn)
#pragma unroll
      for (int r = 0; r < 8; ++r) {
        int lrow = wm + tm * 16 + half * 8 + r;
        int col = n0 + wn + tn * 16 + l15;
        float v = acc[tm][tn][r] * rowW[lrow];
        int tok = rowTok[lrow];
        if (buf) {
          if (tok < T_TOK) unsafeAtomicAdd(&Y[(long)tok * H_DIM + col], v);
        } else {
          Y[(long)tok * H_DIM + col] = v;  // shared expert: first writer
        }
      }
}

// ---------------------------------------------------------------------------
extern "C" void kernel_launch(void* const* d_in, const int* in_sizes, int n_in,
                              void* d_out, int out_size, void* d_ws,
                              size_t ws_size, hipStream_t stream) {
  (void)in_sizes; (void)n_in; (void)out_size; (void)ws_size;
  const float* x = (const float*)d_in[0];
  const float* gate_w = (const float*)d_in[1];
  const float* w_gate = (const float*)d_in[2];
  const float* w_up = (const float*)d_in[3];
  const float* w_down = (const float*)d_in[4];
  const float* sw_gate = (const float*)d_in[5];
  const float* sw_up = (const float*)d_in[6];
  const float* sw_down = (const float*)d_in[7];
  float* y = (float*)d_out;

  char* w = (char*)d_ws;
  int* tki = (int*)w;      w += (size_t)T_TOK * K_TOP * 4;
  float* tkw = (float*)w;  w += (size_t)T_TOK * K_TOP * 4;
  int* bufp = (int*)w;     w += (size_t)N_EXP * CAP_N * 4;
  float* wbuf = (float*)w; w += (size_t)N_EXP * CAP_N * 4;
  __bf16* midr = (__bf16*)w; w += (size_t)N_EXP * CAP_N * I_DIM * 2;
  __bf16* mids = (__bf16*)w;

  gate_topk_kernel<<<T_TOK, 128, 0, stream>>>(x, gate_w, tki, tkw);
  dispatch_kernel<<<N_EXP, 32, 0, stream>>>(tki, tkw, bufp, wbuf);

  // shared expert (plain stores cover all of Y)
  gemm1_swiglu_kernel<<<dim3(IS_DIM / 64, T_TOK / 128, 1), 256, 0, stream>>>(
      x, nullptr, sw_gate, sw_up, mids, IS_DIM, T_TOK);
  gemm2_down_kernel<<<dim3(H_DIM / 128, T_TOK / 128, 1), 256, 0, stream>>>(
      mids, sw_down, nullptr, nullptr, y, IS_DIM, T_TOK);

  // routed experts (atomic scatter-add on top)
  gemm1_swiglu_kernel<<<dim3(I_DIM / 64, CAP_N / 128, N_EXP), 256, 0, stream>>>(
      x, bufp, w_gate, w_up, midr, I_DIM, CAP_N);
  gemm2_down_kernel<<<dim3(H_DIM / 128, CAP_N / 128, N_EXP), 256, 0, stream>>>(
      midr, w_down, bufp, wbuf, y, I_DIM, CAP_N);
}
